// DenseModel_51926154609008
// MI455X (gfx1250) — compile-verified, matrix operand order checked by reading, not run on hardware
//
#include <hip/hip_runtime.h>
#include <hip/hip_bf16.h>

// ---------------------------------------------------------------------------
// Problem constants (fixed by the reference setup_inputs()).
// ---------------------------------------------------------------------------
#define B_Q   2048      // queries
#define DIM   768       // feature dim (K)
#define P_N   16384     // passages (columns of score matrix)
#define NPASS 8         // n_passages -> target col = 8*row

#define MB        64                 // rows per workgroup (4 WMMA row tiles)
#define CSPLIT    8                  // column splits across grid.y
#define COLS_WG   (P_N / CSPLIT)     // 2048 columns per workgroup
#define LOG2E_F   1.4426950408889634f
#define LN2_F     0.6931471805599453f

typedef __bf16 bf16_t;
typedef __attribute__((ext_vector_type(16))) __bf16 v16bf;
typedef __attribute__((ext_vector_type(8)))  float  v8f;

union FragB {
    uint4 u[2];   // two 16-byte runs: K = [8h, 8h+7] then [8h+16, 8h+23]
    v16bf v;
};

// ---------------------------------------------------------------------------
// Kernel 1: split fp32 -> (hi, lo) bf16 pair.  x ~= hi + lo to ~fp32 accuracy.
// ---------------------------------------------------------------------------
__global__ void split_bf16_kernel(const float* __restrict__ src,
                                  bf16_t* __restrict__ hi,
                                  bf16_t* __restrict__ lo, int n) {
    int i = blockIdx.x * blockDim.x + threadIdx.x;
    if (i < n) {
        float x = src[i];
        bf16_t h = (bf16_t)x;          // RTNE
        hi[i] = h;
        lo[i] = (bf16_t)(x - (float)h);
    }
}

// ---------------------------------------------------------------------------
// Kernel 2: exact fp32 target score  s_tgt[i] = dot(q[i], p[8*i]).
// One wave32 per row; shuffle reduction.
// ---------------------------------------------------------------------------
__global__ void stgt_kernel(const float* __restrict__ q,
                            const float* __restrict__ p,
                            float* __restrict__ stgt) {
    int wave = threadIdx.x >> 5;
    int lane = threadIdx.x & 31;
    int row  = blockIdx.x * 8 + wave;
    const float* qr = q + (size_t)row * DIM;
    const float* pr = p + (size_t)row * NPASS * DIM;
    float s = 0.f;
    for (int k = lane; k < DIM; k += 32) s = fmaf(qr[k], pr[k], s);
    #pragma unroll
    for (int m = 16; m >= 1; m >>= 1) s += __shfl_xor(s, m, 32);
    if (lane == 0) stgt[row] = s;
}

// ---------------------------------------------------------------------------
// Kernel 3: fused bf16x3 WMMA GEMM + online softmax (base-2) + rank count.
// grid = (B_Q/MB, CSPLIT), block = 256 threads = 8 waves.
// wave -> (mt = wave&3 row sub-tile of 16, ct = wave>>2 col sub-group of 32).
// Each wave keeps TWO independent accumulator tiles (2 x 16 cols) so the
// matrix pipe sees two interleavable WMMA chains.
// Each WG covers rows [64*bx, 64*bx+64) x cols [2048*by, 2048*by+2048).
// q hi/lo tile: 64x768 bf16 x2 = 192 KB LDS (CDNA5 WGP has 320 KB).
// ---------------------------------------------------------------------------
__global__ __launch_bounds__(256, 1)
void gemm_fused_kernel(const bf16_t* __restrict__ qh, const bf16_t* __restrict__ ql,
                       const bf16_t* __restrict__ ph, const bf16_t* __restrict__ pl,
                       const float* __restrict__ stgt,
                       float* __restrict__ part_m, float* __restrict__ part_s,
                       float* __restrict__ part_c) {
    __shared__ __align__(16) bf16_t ldsQh[MB * DIM];   // 96 KB
    __shared__ __align__(16) bf16_t ldsQl[MB * DIM];   // 96 KB
    __shared__ float statsL[MB][2][3];                 // (m2, s2, cnt) per col sub-group

    const int tid  = threadIdx.x;
    const int lane = tid & 31;
    const int wave = tid >> 5;
    const int mt   = wave & 3;       // row sub-tile (16 rows each), 4 of them
    const int ct   = wave >> 2;      // col sub-group (32 cols each), 2 of them
    const int rowbase = blockIdx.x * MB;
    const int cs      = blockIdx.y;
    const int colsplitbase = cs * COLS_WG;

    // ---- stage q hi/lo tile into LDS (contiguous 64x768 bf16 each) ----
    {
        const uint4* gh = reinterpret_cast<const uint4*>(qh + (size_t)rowbase * DIM);
        const uint4* gl = reinterpret_cast<const uint4*>(ql + (size_t)rowbase * DIM);
        uint4* sh = reinterpret_cast<uint4*>(ldsQh);
        uint4* sl = reinterpret_cast<uint4*>(ldsQl);
        for (int i = tid; i < MB * DIM / 8; i += 256) { sh[i] = gh[i]; sl[i] = gl[i]; }
    }
    __syncthreads();

    const int half = lane >> 4;      // 0: lanes 0-15, 1: lanes 16-31
    const int l16  = lane & 15;
    const int kh   = 8 * half;       // first K-run offset inside a 32-wide step

    // Per-lane per-row state. C/D layout: VGPR v holds row (16*mt + v + 8*half),
    // lane n = l16 is the column within a 16-col tile.
    float m2[8], s2[8], cnt[8], st[8];
    #pragma unroll
    for (int v = 0; v < 8; ++v) {
        m2[v] = -3.0e38f; s2[v] = 0.f; cnt[v] = 0.f;
        st[v] = stgt[rowbase + 16 * mt + v + 8 * half];
    }

    const int arow = 16 * mt + l16;  // A-fragment row for this lane

    for (int cg = 0; cg < COLS_WG / 64; ++cg) {        // 32 column-group iters
        const int colbase = colsplitbase + cg * 64 + ct * 32;
        const int col0    = colbase + l16;             // B columns (= p rows)
        const int col1    = col0 + 16;
        const bf16_t* pbH0 = ph + (size_t)col0 * DIM;
        const bf16_t* pbL0 = pl + (size_t)col0 * DIM;
        const bf16_t* pbH1 = ph + (size_t)col1 * DIM;
        const bf16_t* pbL1 = pl + (size_t)col1 * DIM;

        // Prefetch next column-group's B rows into cache (global_prefetch_b8).
        if (cg + 1 < COLS_WG / 64) {
            __builtin_prefetch(pbH0 + 64 * DIM, 0, 1);
            __builtin_prefetch(pbL0 + 64 * DIM, 0, 1);
            __builtin_prefetch(pbH1 + 64 * DIM, 0, 1);
            __builtin_prefetch(pbL1 + 64 * DIM, 0, 1);
        }

        v8f acc0 = {0.f, 0.f, 0.f, 0.f, 0.f, 0.f, 0.f, 0.f};
        v8f acc1 = {0.f, 0.f, 0.f, 0.f, 0.f, 0.f, 0.f, 0.f};

        #pragma unroll 2
        for (int kk = 0; kk < DIM; kk += 32) {         // 24 K-steps
            FragB aH, aL, bH0, bL0, bH1, bL1;
            const uint4* pa  = reinterpret_cast<const uint4*>(&ldsQh[arow * DIM + kk + kh]);
            const uint4* pal = reinterpret_cast<const uint4*>(&ldsQl[arow * DIM + kk + kh]);
            aH.u[0] = pa[0];  aH.u[1] = pa[2];   // +2 uint4 = +16 bf16 => K-run at +16
            aL.u[0] = pal[0]; aL.u[1] = pal[2];
            const uint4* pb0  = reinterpret_cast<const uint4*>(pbH0 + kk + kh);
            const uint4* pbl0 = reinterpret_cast<const uint4*>(pbL0 + kk + kh);
            const uint4* pb1  = reinterpret_cast<const uint4*>(pbH1 + kk + kh);
            const uint4* pbl1 = reinterpret_cast<const uint4*>(pbL1 + kk + kh);
            bH0.u[0] = pb0[0];  bH0.u[1] = pb0[2];
            bL0.u[0] = pbl0[0]; bL0.u[1] = pbl0[2];
            bH1.u[0] = pb1[0];  bH1.u[1] = pb1[2];
            bL1.u[0] = pbl1[0]; bL1.u[1] = pbl1[2];

            // bf16x3: hi*hi + hi*lo + lo*hi accumulated in fp32.
            // Two independent accumulator chains -> interleavable on the XDL.
            acc0 = __builtin_amdgcn_wmma_f32_16x16x32_bf16(false, aH.v, false, bH0.v,
                                                           (short)0, acc0, false, false);
            acc1 = __builtin_amdgcn_wmma_f32_16x16x32_bf16(false, aH.v, false, bH1.v,
                                                           (short)0, acc1, false, false);
            acc0 = __builtin_amdgcn_wmma_f32_16x16x32_bf16(false, aH.v, false, bL0.v,
                                                           (short)0, acc0, false, false);
            acc1 = __builtin_amdgcn_wmma_f32_16x16x32_bf16(false, aH.v, false, bL1.v,
                                                           (short)0, acc1, false, false);
            acc0 = __builtin_amdgcn_wmma_f32_16x16x32_bf16(false, aL.v, false, bH0.v,
                                                           (short)0, acc0, false, false);
            acc1 = __builtin_amdgcn_wmma_f32_16x16x32_bf16(false, aL.v, false, bH1.v,
                                                           (short)0, acc1, false, false);
        }

        // ---- fused epilogue: online log-sum-exp (base-2) + rank counting ----
        #pragma unroll
        for (int v = 0; v < 8; ++v) {
            int   rowg = rowbase + 16 * mt + v + 8 * half;
            {
                float x  = acc0[v];
                float xm = x * LOG2E_F;
                float nm = fmaxf(m2[v], xm);
                s2[v] = s2[v] * exp2f(m2[v] - nm) + exp2f(xm - nm);
                m2[v] = nm;
                if (x > st[v] && col0 != rowg * NPASS) cnt[v] += 1.f;
            }
            {
                float x  = acc1[v];
                float xm = x * LOG2E_F;
                float nm = fmaxf(m2[v], xm);
                s2[v] = s2[v] * exp2f(m2[v] - nm) + exp2f(xm - nm);
                m2[v] = nm;
                if (x > st[v] && col1 != rowg * NPASS) cnt[v] += 1.f;
            }
        }
    }

    // ---- merge across the 16 lanes sharing each row (butterfly, wave32) ----
    #pragma unroll
    for (int v = 0; v < 8; ++v) {
        #pragma unroll
        for (int mask = 8; mask >= 1; mask >>= 1) {
            float om = __shfl_xor(m2[v], mask, 32);
            float os = __shfl_xor(s2[v], mask, 32);
            float oc = __shfl_xor(cnt[v], mask, 32);
            float nm = fmaxf(m2[v], om);
            s2[v] = s2[v] * exp2f(m2[v] - nm) + os * exp2f(om - nm);
            m2[v] = nm;
            cnt[v] += oc;
        }
        if (l16 == 0) {  // lanes 0 and 16 carry rows v and v+8 of this sub-tile
            int rl = 16 * mt + v + 8 * half;
            statsL[rl][ct][0] = m2[v];
            statsL[rl][ct][1] = s2[v];
            statsL[rl][ct][2] = cnt[v];
        }
    }
    __syncthreads();

    // ---- merge the 2 column sub-groups, emit per-(row, split) partial ----
    if (tid < MB) {
        float m = -3.0e38f, s = 0.f, c = 0.f;
        #pragma unroll
        for (int j = 0; j < 2; ++j) {
            float om = statsL[tid][j][0];
            float os = statsL[tid][j][1];
            float nm = fmaxf(m, om);
            s = s * exp2f(m - nm) + os * exp2f(om - nm);
            m = nm;
            c += statsL[tid][j][2];
        }
        size_t idx = (size_t)(rowbase + tid) * CSPLIT + cs;
        part_m[idx] = m; part_s[idx] = s; part_c[idx] = c;
    }
}

// ---------------------------------------------------------------------------
// Kernel 4: merge splits per row, weighted CE, mean. Single block.
// ---------------------------------------------------------------------------
__global__ void finalize_kernel(const float* __restrict__ pm, const float* __restrict__ ps,
                                const float* __restrict__ pc, const float* __restrict__ stgt,
                                float* __restrict__ out) {
    __shared__ float red[256];
    int tid = threadIdx.x;
    float local = 0.f;
    for (int r = tid; r < B_Q; r += 256) {
        float m = -3.0e38f, s = 0.f, c = 0.f;
        #pragma unroll
        for (int j = 0; j < CSPLIT; ++j) {
            size_t idx = (size_t)r * CSPLIT + j;
            float om = pm[idx], os = ps[idx];
            float nm = fmaxf(m, om);
            s = s * exp2f(m - nm) + os * exp2f(om - nm);
            m = nm;
            c += pc[idx];
        }
        float lse = LN2_F * (m + log2f(s));   // natural-log logsumexp
        float raw = lse - stgt[r];            // per-sample CE
        float d   = c - 1.0f;                 // rank - OPTIMAL_RANK
        float w   = 1.0f + 2.6f * expf(-d * d / 6.48f);  // 2*SIGMA^2 = 6.48
        local += raw * w;
    }
    red[tid] = local;
    __syncthreads();
    for (int sft = 128; sft >= 1; sft >>= 1) {
        if (tid < sft) red[tid] += red[tid + sft];
        __syncthreads();
    }
    if (tid == 0) out[0] = red[0] / (float)B_Q;
}

// ---------------------------------------------------------------------------
// Launch
// ---------------------------------------------------------------------------
extern "C" void kernel_launch(void* const* d_in, const int* in_sizes, int n_in,
                              void* d_out, int out_size, void* d_ws, size_t ws_size,
                              hipStream_t stream) {
    (void)in_sizes; (void)n_in; (void)out_size; (void)ws_size;
    const float* q = (const float*)d_in[0];   // [2048, 768] fp32
    const float* p = (const float*)d_in[1];   // [16384, 768] fp32
    // d_in[2] = n_passages (==8, baked into NPASS)

    char* ws = (char*)d_ws;
    size_t off = 0;
    auto alloc = [&](size_t bytes) -> void* {
        void* r = ws + off;
        off += (bytes + 255) & ~(size_t)255;
        return r;
    };
    bf16_t* qh = (bf16_t*)alloc((size_t)B_Q * DIM * sizeof(bf16_t));
    bf16_t* ql = (bf16_t*)alloc((size_t)B_Q * DIM * sizeof(bf16_t));
    bf16_t* ph = (bf16_t*)alloc((size_t)P_N * DIM * sizeof(bf16_t));
    bf16_t* pl = (bf16_t*)alloc((size_t)P_N * DIM * sizeof(bf16_t));
    float*  stgt = (float*)alloc((size_t)B_Q * sizeof(float));
    float*  pm   = (float*)alloc((size_t)B_Q * CSPLIT * sizeof(float));
    float*  psum = (float*)alloc((size_t)B_Q * CSPLIT * sizeof(float));
    float*  pcnt = (float*)alloc((size_t)B_Q * CSPLIT * sizeof(float));

    int nq = B_Q * DIM;
    int np = P_N * DIM;
    split_bf16_kernel<<<(nq + 255) / 256, 256, 0, stream>>>(q, qh, ql, nq);
    split_bf16_kernel<<<(np + 255) / 256, 256, 0, stream>>>(p, ph, pl, np);
    stgt_kernel<<<B_Q / 8, 256, 0, stream>>>(q, p, stgt);
    dim3 grid(B_Q / MB, CSPLIT);
    gemm_fused_kernel<<<grid, 256, 0, stream>>>(qh, ql, ph, pl, stgt, pm, psum, pcnt);
    finalize_kernel<<<1, 256, 0, stream>>>(pm, psum, pcnt, stgt, (float*)d_out);
}
